// BG_43765716746846
// MI455X (gfx1250) — matrix operand order repaired and checked by reading, not compiled, not run on hardware
//
#include <hip/hip_runtime.h>
#include <hip/hip_bf16.h>

typedef _Float16 f16;
typedef __attribute__((ext_vector_type(16))) _Float16 v16h;
typedef __attribute__((ext_vector_type(8)))  _Float16 v8h;
typedef __attribute__((ext_vector_type(8)))  float    v8f;

#define T_STEPS 4096
#define D_DIM   256
#define U_DIM   2048
#define A_DIM   64

// ---------------------------------------------------------------------------
// Weight prep (B stored TRANSPOSED: BT[N,K] row-major so WMMA fragments are
// contiguous 128-bit loads).
// BT1[u,d]=Wr-Wi ; BT1[2048+u,d]=Wr+Wi   (f16, [4096,256])
// ---------------------------------------------------------------------------
__global__ void prep_l1(const float* __restrict__ l1Wr, const float* __restrict__ l1Wi,
                        const float* __restrict__ l1br, const float* __restrict__ l1bi,
                        f16* __restrict__ BT1, float* __restrict__ bias) {
  int idx = blockIdx.x * blockDim.x + threadIdx.x;   // 2048*256 threads
  int u = idx >> 8;
  int d = idx & 255;
  float wr = l1Wr[u * D_DIM + d], wi = l1Wi[u * D_DIM + d];
  BT1[u * D_DIM + d]           = (f16)(wr - wi);     // N row u      -> yr
  BT1[(U_DIM + u) * D_DIM + d] = (f16)(wr + wi);     // N row 2048+u -> yi
  if (idx < U_DIM) { bias[idx] = l1br[idx]; bias[U_DIM + idx] = l1bi[idx]; }
}

// WdpT[a, u]=Wr[a,u]; WdpT[a, 2048+u]=-Wi[a,u]; WdpT[64+a, u]=Wi[a,u];
// WdpT[64+a, 2048+u]=Wr[a,u]   (f16, [128, 4096]); same for ip.
__global__ void prep_dpip(const float* __restrict__ dpWr, const float* __restrict__ dpWi,
                          const float* __restrict__ ipWr, const float* __restrict__ ipWi,
                          f16* __restrict__ WdpT, f16* __restrict__ WipT) {
  int idx = blockIdx.x * blockDim.x + threadIdx.x;   // 64*2048 threads
  int a = idx >> 11;
  int u = idx & (U_DIM - 1);
  float r1 = dpWr[a * U_DIM + u], i1 = dpWi[a * U_DIM + u];
  WdpT[a * (2 * U_DIM) + u]                     = (f16)r1;
  WdpT[a * (2 * U_DIM) + U_DIM + u]             = (f16)(-i1);
  WdpT[(64 + a) * (2 * U_DIM) + u]              = (f16)i1;
  WdpT[(64 + a) * (2 * U_DIM) + U_DIM + u]      = (f16)r1;
  float r2 = ipWr[a * U_DIM + u], i2 = ipWi[a * U_DIM + u];
  WipT[a * (2 * U_DIM) + u]                     = (f16)r2;
  WipT[a * (2 * U_DIM) + U_DIM + u]             = (f16)(-i2);
  WipT[(64 + a) * (2 * U_DIM) + u]              = (f16)i2;
  WipT[(64 + a) * (2 * U_DIM) + U_DIM + u]      = (f16)r2;
}

// ---------------------------------------------------------------------------
// WMMA GEMM: C[M,N] = act(A[M,K] @ BT[N,K]^T + bias).
// Wave computes a 16x64 tile (4 accumulators); K stepped by 32.
// A fragment = two contiguous v8h loads; each B fragment = two v8h loads.
// ---------------------------------------------------------------------------
template <bool AF32, bool RELU, bool OUTF16>
__global__ void gemm_wmma(const void* __restrict__ Av, const f16* __restrict__ BT,
                          const float* __restrict__ bias, void* __restrict__ Cv,
                          int M, int N, int K) {
  const int lane = threadIdx.x & 31;
  const int wave = threadIdx.x >> 5;
  const int groupsN = N >> 6;                        // 64-wide N groups
  const int tile = blockIdx.x * 8 + wave;            // wave-uniform
  if (tile >= (M >> 4) * groupsN) return;            // whole wave exits
  const int tm = (tile / groupsN) << 4;
  const int tn = (tile % groupsN) << 6;
  const int half = lane >> 4;
  const int l16  = lane & 15;
  const int am = tm + l16;                           // A row for this lane
  const float* Af = (const float*)Av;
  const f16*   Ah = (const f16*)Av;

  v8f acc0 = {}, acc1 = {}, acc2 = {}, acc3 = {};
  for (int k0 = 0; k0 < K; k0 += 32) {
    v16h a;
    if constexpr (AF32) {
      const float* ab = Af + am * K + k0 + half * 8;
      float4 f0 = *(const float4*)(ab);
      float4 f1 = *(const float4*)(ab + 4);
      float4 f2 = *(const float4*)(ab + 16);
      float4 f3 = *(const float4*)(ab + 20);
      a[0] = (f16)f0.x;  a[1] = (f16)f0.y;  a[2]  = (f16)f0.z;  a[3]  = (f16)f0.w;
      a[4] = (f16)f1.x;  a[5] = (f16)f1.y;  a[6]  = (f16)f1.z;  a[7]  = (f16)f1.w;
      a[8] = (f16)f2.x;  a[9] = (f16)f2.y;  a[10] = (f16)f2.z;  a[11] = (f16)f2.w;
      a[12] = (f16)f3.x; a[13] = (f16)f3.y; a[14] = (f16)f3.z;  a[15] = (f16)f3.w;
    } else {
      const f16* ab = Ah + am * K + k0 + half * 8;
      v8h lo = *(const v8h*)(ab);                    // K = g0 octet
      v8h hi = *(const v8h*)(ab + 16);               // K = g1 octet
#pragma unroll
      for (int i = 0; i < 8; ++i) { a[i] = lo[i]; a[8 + i] = hi[i]; }
    }
#pragma unroll
    for (int j = 0; j < 4; ++j) {
      const f16* bb = BT + (tn + j * 16 + l16) * K + k0 + half * 8;
      v8h b0 = *(const v8h*)(bb);
      v8h b1 = *(const v8h*)(bb + 16);
      v16h b;
#pragma unroll
      for (int i = 0; i < 8; ++i) { b[i] = b0[i]; b[8 + i] = b1[i]; }
      v8f& acc = (j == 0) ? acc0 : (j == 1) ? acc1 : (j == 2) ? acc2 : acc3;
      acc = __builtin_amdgcn_wmma_f32_16x16x32_f16(false, a, false, b, (short)0,
                                                   acc, false, false);
    }
  }

#pragma unroll
  for (int j = 0; j < 4; ++j) {
    v8f& acc = (j == 0) ? acc0 : (j == 1) ? acc1 : (j == 2) ? acc2 : acc3;
    int cn = tn + j * 16 + l16;                      // C col for this lane
    float bb = bias ? bias[cn] : 0.0f;
#pragma unroll
    for (int r = 0; r < 8; ++r) {                    // C/D: m = half*8 + r
      int m = tm + half * 8 + r;
      float val = acc[r] + bb;
      if constexpr (RELU) val = fmaxf(val, 0.0f);
      if constexpr (OUTF16) ((f16*)Cv)[m * N + cn] = (f16)val;
      else                  ((float*)Cv)[m * N + cn] = val;
    }
  }
}

// ---------------------------------------------------------------------------
// Hopf recurrences: thread (osc,u) runs its own 4096-step serial scan.
// X layout [T, 4096]: cols 0..2047 = xr, 2048..4095 = xi.  D1/D2 same layout.
// ---------------------------------------------------------------------------
__global__ void hopf_scan(const f16* __restrict__ X,
                          const float* __restrict__ phi1i, const float* __restrict__ phi2i,
                          const float* __restrict__ om1,   const float* __restrict__ om2,
                          f16* __restrict__ D1, f16* __restrict__ D2) {
  int g = blockIdx.x * blockDim.x + threadIdx.x;     // 0..4095
  int u = g & (U_DIM - 1);
  int osc = g >> 11;
  float phi   = (osc ? phi2i[u] : phi1i[u]) * 0.1f;
  float omega = osc ? om2[u] : om1[u];
  f16* D = osc ? D2 : D1;
  float r = 1.0f;
  float c = __cosf(phi), s = __sinf(phi);
  float xr = (float)X[u];
  float xi = (float)X[U_DIM + u];
  for (int t = 0; t < T_STEPS; ++t) {
    int tn = (t + 1 < T_STEPS) ? t + 1 : t;          // prefetch next forcing
    float xrn = (float)X[tn * (2 * U_DIM) + u];
    float xin = (float)X[tn * (2 * U_DIM) + U_DIM + u];
    float rn = r + 0.01f * (r * (1.0f - r * r) + xr * c + xi * s);
    float pn = phi + 0.01f * (omega + (xi * c - xr * s) / fmaxf(r, 1e-6f));
    c = __cosf(pn);
    s = __sinf(pn);
    D[t * (2 * U_DIM) + u]         = (f16)(rn * c);
    D[t * (2 * U_DIM) + U_DIM + u] = (f16)(rn * s);
    r = rn; phi = pn; xr = xrn; xi = xin;
  }
}

// ---------------------------------------------------------------------------
// Critic: value[t] = |d1[t] @ (cWr + i cWi)| ; one block per t.
// ---------------------------------------------------------------------------
__global__ void critic_kernel(const f16* __restrict__ D1,
                              const float* __restrict__ cWr, const float* __restrict__ cWi,
                              const float* __restrict__ cbr, const float* __restrict__ cbi,
                              float* __restrict__ value) {
  int t = blockIdx.x;
  float srr = 0.f, sri = 0.f, sir = 0.f, sii = 0.f;
  for (int u = threadIdx.x; u < U_DIM; u += 256) {
    float dr = (float)D1[t * (2 * U_DIM) + u];
    float di = (float)D1[t * (2 * U_DIM) + U_DIM + u];
    float wr = cWr[u], wi = cWi[u];
    srr += dr * wr; sri += dr * wi; sir += di * wr; sii += di * wi;
  }
  for (int d = 16; d; d >>= 1) {
    srr += __shfl_down(srr, d, 32); sri += __shfl_down(sri, d, 32);
    sir += __shfl_down(sir, d, 32); sii += __shfl_down(sii, d, 32);
  }
  __shared__ float sm[4][8];
  int wv = threadIdx.x >> 5, ln = threadIdx.x & 31;
  if (ln == 0) { sm[0][wv] = srr; sm[1][wv] = sri; sm[2][wv] = sir; sm[3][wv] = sii; }
  __syncthreads();
  if (threadIdx.x == 0) {
    float a = 0.f, b = 0.f, c = 0.f, d = 0.f;
    for (int i = 0; i < 8; ++i) { a += sm[0][i]; b += sm[1][i]; c += sm[2][i]; d += sm[3][i]; }
    float vr = a - d + cbr[0];
    float vi = b + c + cbi[0];
    value[t] = sqrtf(vr * vr + vi * vi);
  }
}

// lD1/lD2 from value[] (p_value is just value[t-1]); also emit final value.
__global__ void ld_kernel(const float* __restrict__ value,
                          float* __restrict__ lD1, float* __restrict__ lD2,
                          float* __restrict__ out) {
  int t = blockIdx.x * blockDim.x + threadIdx.x;
  if (t >= T_STEPS) return;
  float pv = t ? value[t - 1] : 0.0f;
  float dv = 5.0f * (value[t] - pv);
  lD1[t] = 1.0f / (1.0f + __expf(-dv));
  lD2[t] = 1.0f / (1.0f + __expf(dv));
  if (t == T_STEPS - 1) out[A_DIM] = value[t];       // returned scalar value
}

// ---------------------------------------------------------------------------
// gpi prefix-sum: one wave per column a, wave-inclusive scan over t chunks.
// ---------------------------------------------------------------------------
__global__ void gpi_scan(const float* __restrict__ PREdp, const float* __restrict__ PREip,
                         const float* __restrict__ dpbr, const float* __restrict__ dpbi,
                         const float* __restrict__ ipbr, const float* __restrict__ ipbi,
                         const float* __restrict__ noise,
                         const float* __restrict__ lD1, const float* __restrict__ lD2,
                         float* __restrict__ out) {
  int a = blockIdx.x;
  int lane = threadIdx.x;
  float br1 = dpbr[a], bi1 = dpbi[a], br2 = ipbr[a], bi2 = ipbi[a];
  float g = 0.0f;
  for (int t0 = 0; t0 < T_STEPS; t0 += 32) {
    int t = t0 + lane;
    float l1 = lD1[t], l2 = lD2[t];
    float pdr = fmaxf(l1 * PREdp[t * 128 + a]      + br1, 0.0f);
    float pdi = fmaxf(l1 * PREdp[t * 128 + 64 + a] + bi1, 0.0f);
    float pir = fmaxf(l2 * PREip[t * 128 + a]      + br2, 0.0f);
    float pii = fmaxf(l2 * PREip[t * 128 + 64 + a] + bi2, 0.0f);
    float nz  = noise[t * A_DIM + a] * 2.0f * l2 - l2;
    float dr = pdr - (pir + nz);
    float di = pdi - pii;
    float mag = 0.01f * sqrtf(dr * dr + di * di);
    float v = mag;
    for (int d = 1; d < 32; d <<= 1) { float o = __shfl_up(v, d, 32); if (lane >= d) v += o; }
    out[1 + A_DIM + t * A_DIM + a] = g + v;          // gpi_traj
    g += __shfl(v, 31, 32);
  }
}

__global__ void softmax_kernel(float* __restrict__ out) {
  __shared__ float sh[A_DIM];
  int a = threadIdx.x;
  float g = out[1 + A_DIM + (T_STEPS - 1) * A_DIM + a];
  sh[a] = g;
  __syncthreads();
  float mx = -1e30f;
  for (int i = 0; i < A_DIM; ++i) mx = fmaxf(mx, sh[i]);
  float e = __expf(g - mx);
  __syncthreads();
  sh[a] = e;
  __syncthreads();
  float sum = 0.f;
  for (int i = 0; i < A_DIM; ++i) sum += sh[i];
  out[a] = e / sum;                                  // probs
}

// ---------------------------------------------------------------------------
extern "C" void kernel_launch(void* const* d_in, const int* in_sizes, int n_in,
                              void* d_out, int out_size, void* d_ws, size_t ws_size,
                              hipStream_t stream) {
  const float* state = (const float*)d_in[0];
  const float* phi1  = (const float*)d_in[1];
  const float* phi2  = (const float*)d_in[2];
  const float* noise = (const float*)d_in[3];
  const float* om1   = (const float*)d_in[4];
  const float* om2   = (const float*)d_in[5];
  const float* l1Wr  = (const float*)d_in[6];
  const float* l1Wi  = (const float*)d_in[7];
  const float* l1br  = (const float*)d_in[8];
  const float* l1bi  = (const float*)d_in[9];
  const float* dpWr  = (const float*)d_in[10];
  const float* dpWi  = (const float*)d_in[11];
  const float* dpbr  = (const float*)d_in[12];
  const float* dpbi  = (const float*)d_in[13];
  const float* ipWr  = (const float*)d_in[14];
  const float* ipWi  = (const float*)d_in[15];
  const float* ipbr  = (const float*)d_in[16];
  const float* ipbi  = (const float*)d_in[17];
  const float* cWr   = (const float*)d_in[18];
  const float* cWi   = (const float*)d_in[19];
  const float* cbr   = (const float*)d_in[20];
  const float* cbi   = (const float*)d_in[21];
  float* out = (float*)d_out;

  char* ws = (char*)d_ws;
  size_t o = 0;
  f16*   BT1    = (f16*)(ws + o);   o += (size_t)2 * U_DIM * D_DIM * 2;       // 2 MB
  float* bias1  = (float*)(ws + o); o += (size_t)2 * U_DIM * 4;               // 16 KB
  f16*   Xcomb  = (f16*)(ws + o);   o += (size_t)T_STEPS * 2 * U_DIM * 2;     // 32 MB
  f16*   D1     = (f16*)(ws + o);   o += (size_t)T_STEPS * 2 * U_DIM * 2;     // 32 MB
  f16*   D2     = (f16*)(ws + o);   o += (size_t)T_STEPS * 2 * U_DIM * 2;     // 32 MB
  f16*   WdpT   = (f16*)(ws + o);   o += (size_t)128 * 2 * U_DIM * 2;         // 1 MB
  f16*   WipT   = (f16*)(ws + o);   o += (size_t)128 * 2 * U_DIM * 2;         // 1 MB
  float* PREdp  = (float*)(ws + o); o += (size_t)T_STEPS * 128 * 4;           // 2 MB
  float* PREip  = (float*)(ws + o); o += (size_t)T_STEPS * 128 * 4;           // 2 MB
  float* valueA = (float*)(ws + o); o += (size_t)T_STEPS * 4;
  float* lD1A   = (float*)(ws + o); o += (size_t)T_STEPS * 4;
  float* lD2A   = (float*)(ws + o); o += (size_t)T_STEPS * 4;
  if (o > ws_size) return;  // workspace too small; nothing sane to do

  // 1. pack l1 weights (f16, transposed) for combined real|imag GEMM
  prep_l1<<<2048, 256, 0, stream>>>(l1Wr, l1Wi, l1br, l1bi, BT1, bias1);
  // 2. pack dp/ip complex weight blocks (f16, transposed)
  prep_dpip<<<512, 256, 0, stream>>>(dpWr, dpWi, ipWr, ipWi, WdpT, WipT);
  // 3. X = relu(state @ BT1^T + bias)   [4096,256]x[256,4096], WMMA f16
  gemm_wmma<true, true, true><<<2048, 256, 0, stream>>>(
      (const void*)state, BT1, bias1, (void*)Xcomb, T_STEPS, 2 * U_DIM, D_DIM);
  // 4. per-unit Hopf recurrences (the only serial part) -> d1/d2 trajectories
  hopf_scan<<<128, 32, 0, stream>>>(Xcomb, phi1, phi2, om1, om2, D1, D2);
  // 5. critic magnitude per step
  critic_kernel<<<T_STEPS, 256, 0, stream>>>(D1, cWr, cWi, cbr, cbi, valueA);
  // 6. lD1/lD2 gates (p_value = value[t-1]) + final value output
  ld_kernel<<<16, 256, 0, stream>>>(valueA, lD1A, lD2A, out);
  // 7. hoisted dp/ip linears: PRE = D @ WT^T   [4096,4096]x[4096,128], WMMA f16
  gemm_wmma<false, false, false><<<64, 256, 0, stream>>>(
      (const void*)D1, WdpT, nullptr, (void*)PREdp, T_STEPS, 128, 2 * U_DIM);
  gemm_wmma<false, false, false><<<64, 256, 0, stream>>>(
      (const void*)D2, WipT, nullptr, (void*)PREip, T_STEPS, 128, 2 * U_DIM);
  // 8. gated relu + noise + magnitude + prefix-sum over t -> gpi_traj
  gpi_scan<<<A_DIM, 32, 0, stream>>>(PREdp, PREip, dpbr, dpbi, ipbr, ipbi,
                                     noise, lD1A, lD2A, out);
  // 9. softmax over final gpi -> probs
  softmax_kernel<<<1, A_DIM, 0, stream>>>(out);
}
